// AttentionPool_54717883351315
// MI455X (gfx1250) — matrix operand order repaired and checked by reading, not compiled, not run on hardware
//
#include <hip/hip_runtime.h>
#include <hip/hip_bf16.h>
#include <math.h>

typedef __attribute__((ext_vector_type(16))) _Float16 v16h;
typedef __attribute__((ext_vector_type(8)))  float    v8f;

#define D_DIM    512
#define H_DIM    64
#define BGRAPHS  512
#define KB_COUNT 16                          // 512 / 32 K-steps
#define W1H_ELTS (KB_COUNT * 4 * 32 * 16)    // 32768 halves = 64 KB
#define SCAP     2048                        // LDS score capacity per segment

// ---------------------------------------------------------------------------
// Pack W1 (f32, row-major 512x64) into f16 B-fragments laid out exactly as
// V_WMMA_F32_16X16X32_F16 expects (CDNA5 ISA 7.12.2, 16-bit B 32x16):
//   lane in [0,16):  column N = t*16 + lane,      halves h -> K = kb*32 + (h&7) + (h&8 ? 16 : 0)
//   lane in [16,32): column N = t*16 + (lane-16), halves h -> K = kb*32 + 8 + (h&7) + (h&8 ? 16 : 0)
// Stored as w1h[((kb*4 + t)*32 + lane)*16 + h]: each lane's v16h is 32 B
// contiguous.
// ---------------------------------------------------------------------------
__global__ void w1_pack_kernel(const float* __restrict__ W1,
                               _Float16* __restrict__ w1h) {
  int i = blockIdx.x * blockDim.x + threadIdx.x;
  if (i >= W1H_ELTS) return;
  int h    = i & 15;
  int lane = (i >> 4) & 31;
  int t    = (i >> 9) & 3;
  int kb   = i >> 11;
  int K = kb * 32 + ((lane & 16) ? 8 : 0) + (h & 7) + ((h & 8) ? 16 : 0);
  int n = t * 16 + (lane & 15);
  w1h[i] = (_Float16)W1[K * H_DIM + n];
}

// ---------------------------------------------------------------------------
// Fused kernel: one block (256 thr = 8 waves) per segment.
//   Stage packed W1 to LDS via async global->LDS DMA.
//   Phase 1: WMMA scores for the segment's rows (16-row tiles per wave).
//   Phase 2: deterministic LDS-tree softmax (max, sum, in-place weights).
//   Phase 3: weighted accumulation; x rows re-read while still L2-resident.
// ---------------------------------------------------------------------------
__global__ void __launch_bounds__(256)
fused_pool_kernel(const float* __restrict__ x,
                  const _Float16* __restrict__ w1h,
                  const float* __restrict__ b1,
                  const float* __restrict__ W2,
                  const float* __restrict__ b2,
                  const int* __restrict__ batch,
                  float* __restrict__ out,
                  float* __restrict__ spill,     // fallback score storage
                  int Nn) {
  __shared__ __align__(32) _Float16 Bsh[W1H_ELTS];  // 64 KB packed W1
  __shared__ float ssc[SCAP];                        // 8 KB scores/weights
  __shared__ float red[256];

  const int b    = blockIdx.x;
  const int tid  = threadIdx.x;
  const int wave = tid >> 5;
  const int lane = tid & 31;

  // ---- segment range via binary search (batch is sorted) ----
  int lo = 0, hi = Nn;
  while (lo < hi) { int mid = (lo + hi) >> 1; if (batch[mid] < b) lo = mid + 1; else hi = mid; }
  const int start = lo;
  hi = Nn;
  while (lo < hi) { int mid = (lo + hi) >> 1; if (batch[mid] < b + 1) lo = mid + 1; else hi = mid; }
  const int end = lo;
  const int len = end - start;

  // ---- stage packed W1 into LDS with async global->LDS copies ----
  {
    const uint4* src = (const uint4*)w1h;
    for (int i = tid; i < (W1H_ELTS * 2) / 16; i += 256) {   // 4096 x 16B
      unsigned ldsOff = (unsigned)(size_t)((const char*)Bsh + i * 16);
      const uint4* g = src + i;
      asm volatile("global_load_async_to_lds_b128 %0, %1, off"
                   :: "v"(ldsOff), "v"(g) : "memory");
    }
    asm volatile("s_wait_asynccnt 0x0" ::: "memory");
  }
  __syncthreads();

  // ---- Phase 1: scores (WMMA).  Wave w handles 16-row tiles w, w+8, ... ----
  const int n_lo = lane & 15;
  const int half = (lane >> 4) & 1;
  const int nTilesSeg = (len + 15) >> 4;

  float b1v[4], w2v[4];
#pragma unroll
  for (int t = 0; t < 4; ++t) {
    b1v[t] = b1[t * 16 + n_lo];
    w2v[t] = W2[t * 16 + n_lo];
  }
  const float b2v = b2[0];
  const v16h* wB = (const v16h*)Bsh;

  for (int tt = wave; tt < nTilesSeg; tt += 8) {
    int row = start + tt * 16 + n_lo;
    if (row >= end) row = end - 1;                     // tail tile clamp
    const float* xrow = x + (size_t)row * D_DIM + ((lane & 16) ? 8 : 0);

    v8f acc[4] = {};
    for (int kb = 0; kb < KB_COUNT; ++kb) {
      const float* xr = xrow + kb * 32;
      v16h a;
#pragma unroll
      for (int h = 0; h < 8; ++h) a[h]     = (_Float16)xr[h];       // K base+0..7
#pragma unroll
      for (int h = 0; h < 8; ++h) a[8 + h] = (_Float16)xr[16 + h];  // K base+16..23

      v16h bf0 = wB[(kb * 4 + 0) * 32 + lane];
      v16h bf1 = wB[(kb * 4 + 1) * 32 + lane];
      v16h bf2 = wB[(kb * 4 + 2) * 32 + lane];
      v16h bf3 = wB[(kb * 4 + 3) * 32 + lane];

      acc[0] = __builtin_amdgcn_wmma_f32_16x16x32_f16(false, a, false, bf0, (short)0, acc[0], false, false);
      acc[1] = __builtin_amdgcn_wmma_f32_16x16x32_f16(false, a, false, bf1, (short)0, acc[1], false, false);
      acc[2] = __builtin_amdgcn_wmma_f32_16x16x32_f16(false, a, false, bf2, (short)0, acc[2], false, false);
      acc[3] = __builtin_amdgcn_wmma_f32_16x16x32_f16(false, a, false, bf3, (short)0, acc[3], false, false);
    }

    // tanh + dot(W2): C layout lanes0-15 -> M=j, lanes16-31 -> M=j+8; N = lane&15.
    float r[8];
#pragma unroll
    for (int j = 0; j < 8; ++j) r[j] = 0.0f;
#pragma unroll
    for (int t = 0; t < 4; ++t) {
#pragma unroll
      for (int j = 0; j < 8; ++j)
        r[j] += tanhf(acc[t][j] + b1v[t]) * w2v[t];
    }
#pragma unroll
    for (int m = 1; m < 16; m <<= 1) {
#pragma unroll
      for (int j = 0; j < 8; ++j) r[j] += __shfl_xor(r[j], m, 32);
    }
    if (n_lo == 0) {
      const int idxBase = tt * 16 + half * 8;
#pragma unroll
      for (int j = 0; j < 8; ++j) {
        int idx = idxBase + j;
        if (idx < len) {
          float val = r[j] + b2v;
          if (idx < SCAP) ssc[idx] = val;
          else            spill[start + idx] = val;
        }
      }
    }
  }
  if (len > SCAP) __threadfence();
  __syncthreads();

  // ---- Phase 2: segment softmax (deterministic tree reductions) ----
  float m = -INFINITY;
  for (int i = tid; i < len; i += 256) {
    float s = (i < SCAP) ? ssc[i] : spill[start + i];
    m = fmaxf(m, s);
  }
  red[tid] = m; __syncthreads();
  for (int s = 128; s > 0; s >>= 1) {
    if (tid < s) red[tid] = fmaxf(red[tid], red[tid + s]);
    __syncthreads();
  }
  const float maxv = red[0];
  __syncthreads();

  float ssum = 0.0f;
  for (int i = tid; i < len; i += 256) {
    float s = (i < SCAP) ? ssc[i] : spill[start + i];
    ssum += expf(s - maxv);
  }
  red[tid] = ssum; __syncthreads();
  for (int s = 128; s > 0; s >>= 1) {
    if (tid < s) red[tid] += red[tid + s];
    __syncthreads();
  }
  const float invDen = 1.0f / (red[0] + 1e-16f);
  __syncthreads();

  const int lenC = (len < SCAP) ? len : SCAP;
  for (int i = tid; i < lenC; i += 256)
    ssc[i] = expf(ssc[i] - maxv) * invDen;             // weights in place
  __syncthreads();

  // ---- Phase 3: out[b] = sum_i w_i * x_i  (rows still hot in L2) ----
  float2 acc2 = make_float2(0.0f, 0.0f);
  for (int j = 0; j < len; ++j) {
    const float wgt = (j < SCAP) ? ssc[j]
                                 : expf(spill[start + j] - maxv) * invDen;
    const float2* xr = (const float2*)(x + (size_t)(start + j) * D_DIM);
    if (j + 6 < len)
      __builtin_prefetch(x + (size_t)(start + j + 6) * D_DIM + tid * 2, 0, 0);
    const float2 v = xr[tid];
    acc2.x += wgt * v.x;
    acc2.y += wgt * v.y;
  }
  ((float2*)(out + (size_t)b * D_DIM))[tid] = acc2;    // empty segment -> zeros
}

// ---------------------------------------------------------------------------
extern "C" void kernel_launch(void* const* d_in, const int* in_sizes, int n_in,
                              void* d_out, int out_size, void* d_ws, size_t ws_size,
                              hipStream_t stream) {
  const float* x     = (const float*)d_in[0];
  const float* W1    = (const float*)d_in[1];
  const float* b1    = (const float*)d_in[2];
  const float* W2    = (const float*)d_in[3];
  const float* b2    = (const float*)d_in[4];
  const int*   batch = (const int*)d_in[5];
  const int Nn = in_sizes[5];                 // N nodes

  float* spill = (float*)d_ws;                // Nn floats (fallback scores)
  size_t spillBytes = (((size_t)Nn * sizeof(float)) + 255) & ~(size_t)255;
  _Float16* w1h = (_Float16*)((char*)d_ws + spillBytes);

  hipLaunchKernelGGL(w1_pack_kernel, dim3((W1H_ELTS + 255) / 256), dim3(256), 0, stream,
                     W1, w1h);

  hipLaunchKernelGGL(fused_pool_kernel, dim3(BGRAPHS), dim3(256), 0, stream,
                     x, w1h, b1, W2, b2, batch, (float*)d_out, spill, Nn);
}
